// DiscoveryNet_51256139710701
// MI455X (gfx1250) — matrix operand order repaired and checked by reading, not compiled, not run on hardware
//
#include <hip/hip_runtime.h>
#include <hip/hip_bf16.h>
#include <math.h>

// ---------------- problem constants (from reference) ----------------
#define NNODES  50000
#define NEDGES  1600000
#define NGRAPHS 16
#define KSLOTS  8
#define LATENT  4
#define HID     32

typedef __attribute__((ext_vector_type(16))) __bf16 v16bf;
typedef __attribute__((ext_vector_type(8)))  float  v8f;

__device__ __forceinline__ __bf16 f2bf(float f) {
    unsigned u = __builtin_bit_cast(unsigned, f);
    unsigned r = u + 0x7FFFu + ((u >> 16) & 1u);   // round-to-nearest-even
    unsigned short hs = (unsigned short)(r >> 16);
    return __builtin_bit_cast(__bf16, hs);
}

// single v_rcp_f32 (≈1 ulp) instead of IEEE div_scale/div_fmas/div_fixup chain
__device__ __forceinline__ float frcp(float x) { return __builtin_amdgcn_rcpf(x); }

__device__ __forceinline__ float silu(float v) {
    return v * frcp(1.0f + __expf(-v));
}

// ---------------- workspace layout (floats) ----------------
#define OFF_HSUM 0                         // NNODES*HID
#define OFF_CNT  (NNODES * HID)            // NNODES
#define OFF_POOL (OFF_CNT + NNODES)        // NGRAPHS*KSLOTS*HID = 4096
#define OFF_Z    (OFF_POOL + NGRAPHS*KSLOTS*HID)   // NGRAPHS*KSLOTS*LATENT = 512
#define OFF_S    (OFF_Z + NGRAPHS*KSLOTS*LATENT)   // NNODES*KSLOTS
#define ZERO_CNT OFF_S                     // zero hsum+cnt+pool+z each launch

// ---------------- zero-init scratch ----------------
__global__ void zero_kernel(float* __restrict__ p, int n) {
    int i = blockIdx.x * blockDim.x + threadIdx.x;
    int stride = gridDim.x * blockDim.x;
    for (; i < n; i += stride) p[i] = 0.0f;
}

// ---------------- edge message MLP (WMMA bf16) + scatter-add ----------------
#define WPB 8   // waves per block
__global__ void edge_kernel(const float* __restrict__ x,
                            const float* __restrict__ pos,
                            const int*   __restrict__ eidx,
                            const float* __restrict__ w1, const float* __restrict__ b1,
                            const float* __restrict__ w2, const float* __restrict__ b2,
                            float* __restrict__ hsum, float* __restrict__ cnt) {
    __shared__ __bf16 a2t[WPB][16][HID];   // per-wave layer-2 A restage tile
    __shared__ int    dstt[WPB][16];       // per-wave dst indices of the tile

    const int lane = threadIdx.x & 31;
    const int w    = threadIdx.x >> 5;
    const int hi   = lane >> 4;            // 0: lanes 0-15, 1: lanes 16-31
    const int mlo  = lane & 15;
    const int nwaves = (gridDim.x * blockDim.x) >> 5;
    const int wid    = blockIdx.x * (blockDim.x >> 5) + w;

    // ---- build B fragments (held in VGPRs; weights are tiny & L2-hot) ----
    // B layout (bf16 32x16): lane&15 = N column, element e -> K = (e<8 ? e : e+8) + 8*(lane>>4)
    v16bf b1h0{}, b1h1{}, b2h0{}, b2h1{};
#pragma unroll
    for (int e = 0; e < 16; ++e) {
        int k = (e < 8 ? e : e + 8) + 8 * hi;
        if (k < 9) {                        // layer-1 K padded 9 -> 32
            b1h0[e] = f2bf(w1[k * HID + mlo]);
            b1h1[e] = f2bf(w1[k * HID + mlo + 16]);
        }
        b2h0[e] = f2bf(w2[k * HID + mlo]);
        b2h1[e] = f2bf(w2[k * HID + mlo + 16]);
    }
    const float bias1_0 = b1[mlo], bias1_1 = b1[mlo + 16];
    const float bias2_0 = b2[mlo], bias2_1 = b2[mlo + 16];

    const int* srcI = eidx;
    const int* dstI = eidx + NEDGES;
    const int NT = NEDGES / 16;            // 100000 tiles, exact

    for (int t = wid; t < NT; t += nwaves) {
        const int e  = t * 16 + mlo;
        const int si = srcI[e];
        const int di = dstI[e];
        const float4 xd = *(const float4*)(x + di * 4);
        const float4 xs = *(const float4*)(x + si * 4);
        const float dx = pos[si * 3 + 0] - pos[di * 3 + 0];
        const float dy = pos[si * 3 + 1] - pos[di * 3 + 1];
        const float dz = pos[si * 3 + 2] - pos[di * 3 + 2];
        const float dist = sqrtf(dx * dx + dy * dy + dz * dz);

        if (hi == 0) dstt[w][mlo] = di;

        // ---- A1 fragment directly in registers ----
        // lanes 0-15: elements 0..7 = K0..7 ([x_dst, x_src]); 8..15 = K16..23 (pad)
        // lanes16-31: elements 0..7 = K8..15 (dist at K=8);   8..15 = K24..31 (pad)
        v16bf a1{};
        if (hi == 0) {
            a1[0] = f2bf(xd.x); a1[1] = f2bf(xd.y); a1[2] = f2bf(xd.z); a1[3] = f2bf(xd.w);
            a1[4] = f2bf(xs.x); a1[5] = f2bf(xs.y); a1[6] = f2bf(xs.z); a1[7] = f2bf(xs.w);
        } else {
            a1[0] = f2bf(dist);
        }

        v8f c0 = {}, c1 = {};
        c0 = __builtin_amdgcn_wmma_f32_16x16x32_bf16(false, a1, false, b1h0, (short)0, c0, false, false);
        c1 = __builtin_amdgcn_wmma_f32_16x16x32_bf16(false, a1, false, b1h1, (short)0, c1, false, false);

        // ---- bias + SiLU, restage into layer-2 A layout via LDS ----
#pragma unroll
        for (int r = 0; r < 8; ++r) {
            int row = r + 8 * hi;                     // C layout: M = r + 8*(lane>=16)
            float v0 = silu(c0[r] + bias1_0);         // N = mlo
            float v1 = silu(c1[r] + bias1_1);         // N = mlo+16
            a2t[w][row][mlo]      = f2bf(v0);
            a2t[w][row][mlo + 16] = f2bf(v1);
        }
        asm volatile("s_wait_dscnt 0" ::: "memory");  // same-wave LDS RAW fence

        v16bf a2;
#pragma unroll
        for (int e2 = 0; e2 < 16; ++e2) {
            int k = (e2 < 8 ? e2 : e2 + 8) + 8 * hi;  // A layout K mapping
            a2[e2] = a2t[w][mlo][k];
        }

        v8f d0 = {}, d1 = {};
        d0 = __builtin_amdgcn_wmma_f32_16x16x32_bf16(false, a2, false, b2h0, (short)0, d0, false, false);
        d1 = __builtin_amdgcn_wmma_f32_16x16x32_bf16(false, a2, false, b2h1, (short)0, d1, false, false);

        // ---- scatter-add message into node accumulator ----
#pragma unroll
        for (int r = 0; r < 8; ++r) {
            int row = r + 8 * hi;
            int dn  = dstt[w][row];
            atomicAdd(&hsum[dn * HID + mlo],      d0[r] + bias2_0);
            atomicAdd(&hsum[dn * HID + mlo + 16], d1[r] + bias2_1);
        }
        if (hi == 0) atomicAdd(&cnt[di], 1.0f);
    }
}

// ---------------- per-node: mean, softmax pooling, pooled_h accumulate ----------------
__global__ void node_kernel(const float* __restrict__ hsum, const float* __restrict__ cnt,
                            const int*   __restrict__ batch,
                            const float* __restrict__ pw, const float* __restrict__ pb,
                            float* __restrict__ sbuf, float* __restrict__ pooled) {
    __shared__ float psh[NGRAPHS * KSLOTS * HID];   // 16 KB block-local accumulator
    const int tid = threadIdx.x;
    for (int i = tid; i < NGRAPHS * KSLOTS * HID; i += blockDim.x) psh[i] = 0.0f;
    __syncthreads();

    const int n = blockIdx.x * blockDim.x + tid;
    if (n < NNODES) {
        const float inv = frcp(fmaxf(cnt[n], 1.0f));   // cnt is an exact small integer
        float h[HID];
#pragma unroll
        for (int j = 0; j < HID; j += 4) {
            float4 v = *(const float4*)(hsum + n * HID + j);
            h[j] = v.x * inv; h[j+1] = v.y * inv; h[j+2] = v.z * inv; h[j+3] = v.w * inv;
        }
        float lg[KSLOTS];
#pragma unroll
        for (int k = 0; k < KSLOTS; ++k) lg[k] = pb[k];
#pragma unroll
        for (int j = 0; j < HID; ++j)
#pragma unroll
            for (int k = 0; k < KSLOTS; ++k) lg[k] += h[j] * pw[j * KSLOTS + k];
        float mx = lg[0];
#pragma unroll
        for (int k = 1; k < KSLOTS; ++k) mx = fmaxf(mx, lg[k]);
        float den = 0.0f;
#pragma unroll
        for (int k = 0; k < KSLOTS; ++k) { lg[k] = __expf(lg[k] - mx); den += lg[k]; }
        const float rden = frcp(den);
        const int b = batch[n];
#pragma unroll
        for (int k = 0; k < KSLOTS; ++k) {
            float sk = lg[k] * rden;
            sbuf[n * KSLOTS + k] = sk;
            float* dstp = &psh[(b * KSLOTS + k) * HID];
#pragma unroll
            for (int j = 0; j < HID; ++j) atomicAdd(&dstp[j], sk * h[j]);
        }
    }
    __syncthreads();
    for (int i = tid; i < NGRAPHS * KSLOTS * HID; i += blockDim.x) {
        float v = psh[i];
        if (v != 0.0f) atomicAdd(&pooled[i], v);
    }
}

// ---------------- z = pooled @ toz_w + toz_b ----------------
__global__ void z_kernel(const float* __restrict__ pooled,
                         const float* __restrict__ tw, const float* __restrict__ tb,
                         float* __restrict__ z) {
    int t = threadIdx.x;                 // 128 threads == NGRAPHS*KSLOTS
    if (t >= NGRAPHS * KSLOTS) return;
    const float* p = pooled + t * HID;
#pragma unroll
    for (int d = 0; d < LATENT; ++d) {
        float acc = tb[d];
#pragma unroll
        for (int j = 0; j < HID; ++j) acc += p[j] * tw[j * LATENT + d];
        z[t * LATENT + d] = acc;
    }
}

// ---------------- decode: q = s·z ; silu(q@W1+b1)@W2+b2 ----------------
__global__ void decode_kernel(const float* __restrict__ sbuf, const float* __restrict__ z,
                              const float* __restrict__ dw1, const float* __restrict__ db1,
                              const float* __restrict__ dw2, const float* __restrict__ db2,
                              float* __restrict__ out) {
    __shared__ float zsh[KSLOTS * LATENT];
    const int b = blockIdx.y;
    if (threadIdx.x < KSLOTS * LATENT) zsh[threadIdx.x] = z[b * KSLOTS * LATENT + threadIdx.x];
    __syncthreads();
    const int n = blockIdx.x * blockDim.x + threadIdx.x;
    if (n >= NNODES) return;

    float4 s0 = *(const float4*)(sbuf + n * KSLOTS);
    float4 s1 = *(const float4*)(sbuf + n * KSLOTS + 4);
    float sv[KSLOTS] = { s0.x, s0.y, s0.z, s0.w, s1.x, s1.y, s1.z, s1.w };

    float q[LATENT] = {0.f, 0.f, 0.f, 0.f};
#pragma unroll
    for (int k = 0; k < KSLOTS; ++k)
#pragma unroll
        for (int d = 0; d < LATENT; ++d) q[d] += sv[k] * zsh[k * LATENT + d];

    float o[LATENT] = { db2[0], db2[1], db2[2], db2[3] };
#pragma unroll
    for (int j = 0; j < HID; ++j) {
        float a = db1[j];
#pragma unroll
        for (int d = 0; d < LATENT; ++d) a += q[d] * dw1[d * HID + j];
        a = silu(a);
#pragma unroll
        for (int d = 0; d < LATENT; ++d) o[d] += a * dw2[j * LATENT + d];
    }
    float4 ov = { o[0], o[1], o[2], o[3] };
    *(float4*)(out + ((size_t)b * NNODES + n) * 4) = ov;
}

// ---------------- launch ----------------
extern "C" void kernel_launch(void* const* d_in, const int* in_sizes, int n_in,
                              void* d_out, int out_size, void* d_ws, size_t ws_size,
                              hipStream_t stream) {
    (void)in_sizes; (void)n_in; (void)out_size; (void)ws_size;
    const float* x      = (const float*)d_in[0];
    const float* pos    = (const float*)d_in[1];
    const int*   eidx   = (const int*)  d_in[2];
    const int*   batch  = (const int*)  d_in[3];
    const float* enc_w1 = (const float*)d_in[4];
    const float* enc_b1 = (const float*)d_in[5];
    const float* enc_w2 = (const float*)d_in[6];
    const float* enc_b2 = (const float*)d_in[7];
    const float* pool_w = (const float*)d_in[8];
    const float* pool_b = (const float*)d_in[9];
    const float* toz_w  = (const float*)d_in[10];
    const float* toz_b  = (const float*)d_in[11];
    const float* dec_w1 = (const float*)d_in[12];
    const float* dec_b1 = (const float*)d_in[13];
    const float* dec_w2 = (const float*)d_in[14];
    const float* dec_b2 = (const float*)d_in[15];
    float* out = (float*)d_out;
    float* ws  = (float*)d_ws;

    float* hsum   = ws + OFF_HSUM;
    float* cnt    = ws + OFF_CNT;
    float* pooled = ws + OFF_POOL;
    float* zb     = ws + OFF_Z;
    float* sbuf   = ws + OFF_S;

    zero_kernel<<<1024, 256, 0, stream>>>(ws, ZERO_CNT);

    edge_kernel<<<1024, 256, 0, stream>>>(x, pos, eidx, enc_w1, enc_b1, enc_w2, enc_b2,
                                          hsum, cnt);

    node_kernel<<<(NNODES + 255) / 256, 256, 0, stream>>>(hsum, cnt, batch, pool_w, pool_b,
                                                          sbuf, pooled);

    z_kernel<<<1, 128, 0, stream>>>(pooled, toz_w, toz_b, zb);

    dim3 dgrid((NNODES + 255) / 256, NGRAPHS);
    decode_kernel<<<dgrid, 256, 0, stream>>>(sbuf, zb, dec_w1, dec_b1, dec_w2, dec_b2, out);
}
